// Cross_attention2_63041529971111
// MI455X (gfx1250) — matrix operand order repaired and checked by reading, not compiled, not run on hardware
//
#include <hip/hip_runtime.h>
#include <hip/hip_bf16.h>

typedef __bf16 bf16;
typedef __attribute__((ext_vector_type(16))) __bf16 v16bf;
typedef __attribute__((ext_vector_type(8)))  float  v8f;
typedef __attribute__((ext_vector_type(4))) unsigned int u32x4;
typedef __attribute__((ext_vector_type(8))) int i32x8;
typedef __attribute__((ext_vector_type(4))) int i32x4;

#define BATCH 16
#define LSEQ  512
#define DIM   512
#define NH    8
#define ODIM  (NH * DIM)   // 4096

// Padded LDS row strides (in bf16 elements) chosen for conflict-free ds_load_b128:
// K tile rows: 512 + 8  -> 1040 B -> bank stride 4  (16 lanes tile all 64 banks)
// V tile rows: 32 + 8   -> 80 B   -> bank stride 20 (16 lanes tile all 64 banks)
#define KROW 520
#define VROW 40

__device__ __forceinline__ v8f wmma_bf16(v16bf a, v16bf b, v8f c) {
  return __builtin_amdgcn_wmma_f32_16x16x32_bf16(
      false, a, false, b, (short)0, c, false, false);
}

__device__ __forceinline__ v16bf load_frag2(const bf16* p0, const bf16* p1) {
  union { uint4 u[2]; v16bf v; } t;
  t.u[0] = *(const uint4*)p0;
  t.u[1] = *(const uint4*)p1;
  return t.v;
}

__device__ __forceinline__ v16bf cvt_frag(const float* r0, const float* r1) {
  float4 a = ((const float4*)r0)[0], b = ((const float4*)r0)[1];
  float4 c = ((const float4*)r1)[0], d = ((const float4*)r1)[1];
  v16bf v;
  v[0]=(bf16)a.x;  v[1]=(bf16)a.y;  v[2]=(bf16)a.z;  v[3]=(bf16)a.w;
  v[4]=(bf16)b.x;  v[5]=(bf16)b.y;  v[6]=(bf16)b.z;  v[7]=(bf16)b.w;
  v[8]=(bf16)c.x;  v[9]=(bf16)c.y;  v[10]=(bf16)c.z; v[11]=(bf16)c.w;
  v[12]=(bf16)d.x; v[13]=(bf16)d.y; v[14]=(bf16)d.z; v[15]=(bf16)d.w;
  return v;
}

__device__ __forceinline__ float halfmax(float v) {
  v = fmaxf(v, __shfl_xor(v, 1, 16));
  v = fmaxf(v, __shfl_xor(v, 2, 16));
  v = fmaxf(v, __shfl_xor(v, 4, 16));
  v = fmaxf(v, __shfl_xor(v, 8, 16));
  return v;
}
__device__ __forceinline__ float halfsum(float v) {
  v += __shfl_xor(v, 1, 16);
  v += __shfl_xor(v, 2, 16);
  v += __shfl_xor(v, 4, 16);
  v += __shfl_xor(v, 8, 16);
  return v;
}

// Issue a 2D TDM load (global -> LDS) with optional LDS padding.
// Bitfields per CDNA5 ISA 8.3/8.4: group0 {count=1, lds_addr, global_addr, type=2},
// group1 {data_size=2B, pad_enable, pad codes, tensor/tile dims, dim0 stride}.
// This toolchain's builtin takes 6 args (g0, g1, g2, g3, extra-group, cpol).
__device__ __forceinline__ void tdm_load_2d(unsigned lds_addr,
                                            unsigned long long gaddr,
                                            unsigned tile_d0, unsigned tile_d1,
                                            unsigned tensor_d0, unsigned tensor_d1,
                                            unsigned stride0,
                                            unsigned pad_interval_code,
                                            unsigned pad_amount_code) {
  u32x4 g0;
  g0[0] = 1u;                                        // count=1 (valid descriptor)
  g0[1] = lds_addr;                                  // LDS byte address
  g0[2] = (unsigned)(gaddr & 0xffffffffull);         // global addr [31:0]
  g0[3] = (unsigned)((gaddr >> 32) & 0x1ffffffull)   // global addr [56:32]
        | (2u << 30);                                // type = 2 ("image")
  i32x8 g1;
  g1[0] = (int)((1u << 16)                           // data_size = 1 -> 2 bytes
              | (1u << 20)                           // pad_enable
              | (pad_interval_code << 22)
              | (pad_amount_code << 25));
  g1[1] = (int)((tensor_d0 & 0xffffu) << 16);
  g1[2] = (int)(((tensor_d0 >> 16) & 0xffffu) | ((tensor_d1 & 0xffffu) << 16));
  g1[3] = (int)(((tensor_d1 >> 16) & 0xffffu) | ((tile_d0 & 0xffffu) << 16));
  g1[4] = (int)(tile_d1 & 0xffffu);                  // tile_dim1 (tile_dim2 = 0)
  g1[5] = (int)stride0;                              // tensor_dim0_stride[31:0]
  g1[6] = 0;                                         // stride0[47:32], stride1[15:0]
  g1[7] = 0;
  i32x4 z4 = {0, 0, 0, 0};
  i32x8 z8 = {0, 0, 0, 0, 0, 0, 0, 0};
  __builtin_amdgcn_tensor_load_to_lds(g0, g1, z4, z4, z8, 0);
}

// ---------------- Kernel 0: W (h,d,e) f32 -> Wt (h,e,d) bf16 ----------------
__global__ __launch_bounds__(256) void k_prep_w(const float* __restrict__ W,
                                                bf16* __restrict__ Wt) {
  int idx = blockIdx.x * 256 + threadIdx.x;
  int d = idx & (DIM - 1);
  int e = (idx >> 9) & (DIM - 1);
  int h = idx >> 18;
  Wt[idx] = (bf16)W[((size_t)(h << 9) + d) * DIM + e];
}

// ---------------- Kernel 1: projections Q,K (row-major) and V (transposed) ----------------
__global__ __launch_bounds__(256) void k_proj(
    const float* __restrict__ Xq, const float* __restrict__ Xk, const float* __restrict__ Xv,
    const bf16* __restrict__ Wt, const float* __restrict__ bias,
    bf16* __restrict__ Qp, bf16* __restrict__ Kp, bf16* __restrict__ Vt) {
  const int lane = threadIdx.x & 31;
  const int w    = threadIdx.x >> 5;
  const int wm   = w >> 1, wn = w & 1;
  const int src  = blockIdx.z >> 7;
  const int bh   = blockIdx.z & 127;
  const int b    = bh >> 3, h = bh & 7;
  const int m0   = blockIdx.x * 128 + wm * 32;
  const int n0   = blockIdx.y * 64  + wn * 32;
  const int ln   = lane & 15;
  const int hi   = lane >> 4;

  const float* X  = (src == 0) ? Xq : (src == 1) ? Xk : Xv;
  const float* Xb = X + (size_t)b * LSEQ * DIM;
  const bf16*  Wb = Wt + (size_t)h * DIM * DIM;

  v8f acc[2][2] = {};

  #pragma unroll 4
  for (int c = 0; c < 16; ++c) {
    const int ch = c * 32;
    v16bf a[2], bb[2];
    #pragma unroll
    for (int mb = 0; mb < 2; ++mb) {
      const float* r = Xb + (size_t)(m0 + mb * 16 + ln) * DIM + ch + hi * 8;
      a[mb] = cvt_frag(r, r + 16);
    }
    #pragma unroll
    for (int nb = 0; nb < 2; ++nb) {
      const bf16* p = Wb + (size_t)(n0 + nb * 16 + ln) * DIM + ch + hi * 8;
      bb[nb] = load_frag2(p, p + 16);
    }
    #pragma unroll
    for (int mb = 0; mb < 2; ++mb)
      #pragma unroll
      for (int nb = 0; nb < 2; ++nb)
        acc[mb][nb] = wmma_bf16(a[mb], bb[nb], acc[mb][nb]);
  }

  if (src < 2) {
    bf16* Out = (src == 0 ? Qp : Kp) + (size_t)bh * LSEQ * DIM;
    #pragma unroll
    for (int mb = 0; mb < 2; ++mb)
      #pragma unroll
      for (int nb = 0; nb < 2; ++nb) {
        const int e = n0 + nb * 16 + ln;
        const float bv = bias[h * DIM + e];
        #pragma unroll
        for (int r = 0; r < 8; ++r) {
          const int l = m0 + mb * 16 + r + hi * 8;
          Out[(size_t)l * DIM + e] = (bf16)(acc[mb][nb][r] + bv);
        }
      }
  } else {
    bf16* Out = Vt + (size_t)bh * DIM * LSEQ;        // [d][l]
    #pragma unroll
    for (int mb = 0; mb < 2; ++mb)
      #pragma unroll
      for (int nb = 0; nb < 2; ++nb) {
        const int e = n0 + nb * 16 + ln;
        const float bv = bias[h * DIM + e];
        union { bf16 x[8]; uint4 u; } pk;
        #pragma unroll
        for (int r = 0; r < 8; ++r) pk.x[r] = (bf16)(acc[mb][nb][r] + bv);
        const int l0 = m0 + mb * 16 + hi * 8;
        *(uint4*)&Out[(size_t)e * LSEQ + l0] = pk.u;
      }
  }
}

// ---------------- Kernel 2: fused flash attention with TDM-staged K/V ----------------
__global__ __launch_bounds__(512) void k_attn(
    const bf16* __restrict__ Qp, const bf16* __restrict__ Kp, const bf16* __restrict__ Vt,
    const int* __restrict__ mask, float* __restrict__ out) {
  __shared__ bf16  Kt[2][32 * KROW];     // 2 x 33,280 B  (rows padded +16B)
  __shared__ bf16  Vtile[2][512 * VROW]; // 2 x 40,960 B  (rows padded +16B)
  __shared__ float Sred[4][4][16 * 32];  // 32 KB
  __shared__ bf16  Pbuf[4][16 * 32];     // 4 KB

  const int lane = threadIdx.x & 31;
  const int w    = threadIdx.x >> 5;
  const int wq   = w >> 2, wd = w & 3;
  const int ln   = lane & 15, hi = lane >> 4;
  const int bh   = blockIdx.y, b = bh >> 3, h = bh & 7;
  const int q0   = blockIdx.x * 64 + wq * 16;
  const int d0   = wd * 128;

  const bf16* Qb = Qp + (size_t)bh * LSEQ * DIM;
  const bf16* Kb = Kp + (size_t)bh * LSEQ * DIM;
  const bf16* Vb = Vt + (size_t)bh * DIM * LSEQ;

  // Resident Q A-fragments (16 x 128 slice)
  v16bf Aq[4];
  #pragma unroll
  for (int c = 0; c < 4; ++c) {
    const bf16* p = Qb + (size_t)(q0 + ln) * DIM + d0 + c * 32 + hi * 8;
    Aq[c] = load_frag2(p, p + 16);
  }

  v8f O[8] = {};
  float m[8], l[8];
  #pragma unroll
  for (int r = 0; r < 8; ++r) { m[r] = -INFINITY; l[r] = 0.0f; }

  const float inv512 = 1.0f / 512.0f;
  const float negc   = -1.0e9f * 0.044194173824159216f;  // NEG / sqrt(512)

  // Prologue: wave 0 DMAs tile 0 (K: 32x512 rows, V: 512x32 cols) into buffer 0.
  if (w == 0) {
    tdm_load_2d((unsigned)(uintptr_t)&Kt[0][0],
                (unsigned long long)(uintptr_t)Kb,
                /*tile*/ 512, 32, /*tensor*/ 512, 512, /*stride0*/ 512,
                /*pad: every 256 dw (1024B) add 4 dw*/ 7, 3);
    tdm_load_2d((unsigned)(uintptr_t)&Vtile[0][0],
                (unsigned long long)(uintptr_t)Vb,
                /*tile*/ 32, 512, /*tensor*/ 512, 512, /*stride0*/ 512,
                /*pad: every 16 dw (64B) add 4 dw*/ 3, 3);
  }

  for (int kt = 0; kt < 16; ++kt) {
    const int kb  = kt * 32;
    const int buf = kt & 1;
    __syncthreads();   // everyone done reading buf^1 (previous iteration)

    if (w == 0) {
      if (kt + 1 < 16) {   // prefetch next tile into the other buffer
        const int nkb = kb + 32;
        tdm_load_2d((unsigned)(uintptr_t)&Kt[buf ^ 1][0],
                    (unsigned long long)(uintptr_t)(Kb + (size_t)nkb * DIM),
                    512, 32, 512, 512, 512, 7, 3);
        tdm_load_2d((unsigned)(uintptr_t)&Vtile[buf ^ 1][0],
                    (unsigned long long)(uintptr_t)(Vb + nkb),
                    32, 512, 512, 512, 512, 3, 3);
        __builtin_amdgcn_s_wait_tensorcnt((short)2);  // current tile complete
      } else {
        __builtin_amdgcn_s_wait_tensorcnt((short)0);
      }
    }
    __syncthreads();   // current K/V tile visible to all waves

    // Partial S over this wave's 128-wide d-slice (from LDS, bank-conflict-free)
    v8f Sc[2] = {};
    #pragma unroll
    for (int c = 0; c < 4; ++c)
      #pragma unroll
      for (int nb = 0; nb < 2; ++nb) {
        const bf16* p = &Kt[buf][(nb * 16 + ln) * KROW + d0 + c * 32 + hi * 8];
        Sc[nb] = wmma_bf16(Aq[c], load_frag2(p, p + 16), Sc[nb]);
      }

    #pragma unroll
    for (int nb = 0; nb < 2; ++nb)
      #pragma unroll
      for (int r = 0; r < 8; ++r)
        Sred[wq][wd][(r + hi * 8) * 32 + nb * 16 + ln] = Sc[nb][r];
    __syncthreads();

    // Sum partials, scale, mask, online softmax
    float p0[8], p1[8];
    #pragma unroll
    for (int r = 0; r < 8; ++r) {
      const int M = r + hi * 8;
      const int q = q0 + M;
      float s0 = 0.0f, s1 = 0.0f;
      #pragma unroll
      for (int t = 0; t < 4; ++t) {
        s0 += Sred[wq][t][M * 32 + ln];
        s1 += Sred[wq][t][M * 32 + 16 + ln];
      }
      const int mv0 = mask[((size_t)b * LSEQ + q) * LSEQ + kb + ln];
      const int mv1 = mask[((size_t)b * LSEQ + q) * LSEQ + kb + 16 + ln];
      s0 = mv0 ? s0 * inv512 : negc;
      s1 = mv1 ? s1 * inv512 : negc;
      const float rx   = halfmax(fmaxf(s0, s1));
      const float mnew = fmaxf(m[r], rx);
      p0[r] = __expf(s0 - mnew);
      p1[r] = __expf(s1 - mnew);
      const float rs = halfsum(p0[r] + p1[r]);
      const float sc = __expf(m[r] - mnew);
      l[r] = l[r] * sc + rs;
      m[r] = mnew;
      #pragma unroll
      for (int f = 0; f < 8; ++f) O[f][r] *= sc;
    }

    if (wd == 0) {
      #pragma unroll
      for (int r = 0; r < 8; ++r) {
        const int M = r + hi * 8;
        Pbuf[wq][M * 32 + ln]      = (bf16)p0[r];
        Pbuf[wq][M * 32 + 16 + ln] = (bf16)p1[r];
      }
    }
    __syncthreads();

    // O += P * V_slice (P re-read in A layout; V from padded LDS tile)
    const bf16* pp = &Pbuf[wq][ln * 32 + hi * 8];
    v16bf Pa = load_frag2(pp, pp + 16);
    #pragma unroll
    for (int f = 0; f < 8; ++f) {
      const bf16* vp = &Vtile[buf][(d0 + f * 16 + ln) * VROW + hi * 8];
      O[f] = wmma_bf16(Pa, load_frag2(vp, vp + 16), O[f]);
    }
  }

  // Normalize and store out[b, q, h*D + d]
  #pragma unroll
  for (int f = 0; f < 8; ++f) {
    const int dcol = d0 + f * 16 + ln;
    #pragma unroll
    for (int r = 0; r < 8; ++r) {
      const int q = q0 + r + hi * 8;
      out[((size_t)b * LSEQ + q) * ODIM + h * DIM + dcol] = O[f][r] / l[r];
    }
  }
}

extern "C" void kernel_launch(void* const* d_in, const int* in_sizes, int n_in,
                              void* d_out, int out_size, void* d_ws, size_t ws_size,
                              hipStream_t stream) {
  (void)in_sizes; (void)n_in; (void)out_size; (void)ws_size;
  const float* query = (const float*)d_in[0];
  const float* key   = (const float*)d_in[1];
  const float* value = (const float*)d_in[2];
  const int*   mask  = (const int*)d_in[3];
  const float* Wq    = (const float*)d_in[4];
  const float* bq    = (const float*)d_in[5];
  float* out = (float*)d_out;

  char* ws = (char*)d_ws;
  size_t off = 0;
  bf16* Wt = (bf16*)(ws + off); off += (size_t)NH * DIM * DIM * sizeof(bf16);
  bf16* Qp = (bf16*)(ws + off); off += (size_t)BATCH * NH * LSEQ * DIM * sizeof(bf16);
  bf16* Kp = (bf16*)(ws + off); off += (size_t)BATCH * NH * LSEQ * DIM * sizeof(bf16);
  bf16* Vt = (bf16*)(ws + off); off += (size_t)BATCH * NH * LSEQ * DIM * sizeof(bf16);

  hipLaunchKernelGGL(k_prep_w, dim3((NH * DIM * DIM) / 256), dim3(256), 0, stream, Wq, Wt);
  hipLaunchKernelGGL(k_proj, dim3(4, 8, 3 * BATCH * NH), dim3(256), 0, stream,
                     query, key, value, Wt, bq, Qp, Kp, Vt);
  hipLaunchKernelGGL(k_attn, dim3(LSEQ / 64, BATCH * NH), dim3(512), 0, stream,
                     Qp, Kp, Vt, mask, out);
}
